// ScaledDotProductAttention_44942537785819
// MI455X (gfx1250) — compile-verified, matrix operand order checked by reading, not run on hardware
//
#include <hip/hip_runtime.h>
#include <math.h>

typedef __attribute__((ext_vector_type(16))) __bf16        v16bf;
typedef __attribute__((ext_vector_type(8)))  float         v8f;
typedef __attribute__((ext_vector_type(4)))  float         v4f;
typedef __attribute__((ext_vector_type(4)))  unsigned int  v4u;

namespace {

constexpr int   S_LEN   = 2048;
constexpr int   D_HEAD  = 64;
constexpr int   N_HEAD  = 16;
constexpr int   N_BATCH = 2;
constexpr int   WPB     = 4;          // waves per block (all same head)
constexpr int   NKT     = S_LEN / 32; // key tiles of 32
// softmax runs in log2 domain: fold 1/sqrt(64) * log2(e) into Q
constexpr float QK_SCALE_LOG2E = 0.125f * 1.44269504088896340736f;
constexpr float MASKED = -1.0e30f;    // finite sentinel -> branchless softmax

// padded LDS row strides (ushorts); byte strides 144/80/80 are 16B-aligned
// and conflict-free across 16 lanes.
constexpr int KROW = 72;   // sK: [32 keys][64 d] bf16
constexpr int VROW = 40;   // sV: [64 d][32 keys] bf16 (transposed)
constexpr int PROW = 40;   // sP: per-wave [16 rows][32 keys] bf16

__device__ __forceinline__ float fast_exp2(float x) {
#if __has_builtin(__builtin_amdgcn_exp2f)
  return __builtin_amdgcn_exp2f(x);   // v_exp_f32 (hardware is base-2)
#else
  return exp2f(x);
#endif
}
__device__ __forceinline__ float fast_log2(float x) {
#if __has_builtin(__builtin_amdgcn_logf)
  return __builtin_amdgcn_logf(x);    // v_log_f32 (base-2)
#else
  return log2f(x);
#endif
}

__device__ __forceinline__ unsigned short f2bf_bits(float f) {
  union { __bf16 h; unsigned short u; } c;
  c.h = (__bf16)f;                       // native bf16 cvt on gfx1250
  return c.u;
}
__device__ __forceinline__ unsigned int pack2(float lo, float hi) {
  return (unsigned int)f2bf_bits(lo) | ((unsigned int)f2bf_bits(hi) << 16);
}
__device__ __forceinline__ v4u pack8(v4f a, v4f b) {
  v4u w;
  w.x = pack2(a.x, a.y); w.y = pack2(a.z, a.w);
  w.z = pack2(b.x, b.y); w.w = pack2(b.z, b.w);
  return w;
}

union Frag { v16bf v; unsigned short u[16]; v4u q[2]; };

__device__ __forceinline__ v8f wmma_bf16(v16bf a, v16bf b, v8f c) {
  return __builtin_amdgcn_wmma_f32_16x16x32_bf16(
      false, a, false, b, (short)0, c, false, false);
}

// ---- cooperative staging: 128 threads, one 32x64 fp32 tile -> bf16 LDS ----
__device__ __forceinline__ void stage_K(const float* __restrict__ src,
                                        unsigned short* __restrict__ dst,
                                        int tid) {
  const int r  = tid >> 2;            // key row 0..31
  const int d0 = (tid & 3) << 4;      // 16-float chunk
  const float* s = src + (size_t)r * D_HEAD + d0;   // coalesced b128 loads
  v4f a = *(const v4f*)(s);
  v4f b = *(const v4f*)(s + 4);
  v4f c = *(const v4f*)(s + 8);
  v4f d = *(const v4f*)(s + 12);
  *(v4u*)(dst + r * KROW + d0)     = pack8(a, b);   // ds_store_b128
  *(v4u*)(dst + r * KROW + d0 + 8) = pack8(c, d);
}

__device__ __forceinline__ void stage_VT(const float* __restrict__ src,
                                         unsigned short* __restrict__ dst,
                                         int tid) {
  const int r  = tid >> 2;            // key row 0..31
  const int d0 = (tid & 3) << 4;
  const float* s = src + (size_t)r * D_HEAD + d0;   // coalesced reads
  v4f a = *(const v4f*)(s);
  v4f b = *(const v4f*)(s + 4);
  v4f c = *(const v4f*)(s + 8);
  v4f d = *(const v4f*)(s + 12);
  unsigned short* t = dst + (size_t)d0 * VROW + r;  // transposed scatter
  t[0 * VROW]  = f2bf_bits(a.x); t[1 * VROW]  = f2bf_bits(a.y);
  t[2 * VROW]  = f2bf_bits(a.z); t[3 * VROW]  = f2bf_bits(a.w);
  t[4 * VROW]  = f2bf_bits(b.x); t[5 * VROW]  = f2bf_bits(b.y);
  t[6 * VROW]  = f2bf_bits(b.z); t[7 * VROW]  = f2bf_bits(b.w);
  t[8 * VROW]  = f2bf_bits(c.x); t[9 * VROW]  = f2bf_bits(c.y);
  t[10 * VROW] = f2bf_bits(c.z); t[11 * VROW] = f2bf_bits(c.w);
  t[12 * VROW] = f2bf_bits(d.x); t[13 * VROW] = f2bf_bits(d.y);
  t[14 * VROW] = f2bf_bits(d.z); t[15 * VROW] = f2bf_bits(d.w);
}

// ---- 16x32 masked score tile; K fragments = 2x ds_load_b128 each ----
__device__ __forceinline__ void score_tile(
    const unsigned short* __restrict__ sK, const int* __restrict__ Mb,
    v16bf aq0, v16bf aq1, int n0, int row0, int half, int c16,
    v8f& s0, v8f& s1)
{
  Frag b00, b01, b10, b11;
  const unsigned short* r0 = sK + c16 * KROW;        // key n0+c16
  const unsigned short* r1 = r0 + 16 * KROW;         // key n0+16+c16
  const int o = half << 3;                           // lane-half K offset
  b00.q[0] = *(const v4u*)(r0 + o);       b00.q[1] = *(const v4u*)(r0 + 16 + o);
  b01.q[0] = *(const v4u*)(r0 + 32 + o);  b01.q[1] = *(const v4u*)(r0 + 48 + o);
  b10.q[0] = *(const v4u*)(r1 + o);       b10.q[1] = *(const v4u*)(r1 + 16 + o);
  b11.q[0] = *(const v4u*)(r1 + 32 + o);  b11.q[1] = *(const v4u*)(r1 + 48 + o);
  s0 = wmma_bf16(aq0, b00.v, s0);
  s0 = wmma_bf16(aq1, b01.v, s0);
  s1 = wmma_bf16(aq0, b10.v, s1);
  s1 = wmma_bf16(aq1, b11.v, s1);
#pragma unroll
  for (int j = 0; j < 8; ++j) {
    const int* mp = Mb + (size_t)(row0 + j + 8 * half) * S_LEN + n0 + c16;
    s0[j] = (mp[0]  == 0) ? MASKED : s0[j];   // v_cndmask, no branches
    s1[j] = (mp[16] == 0) ? MASKED : s1[j];
  }
}

__global__ __launch_bounds__(WPB * 32)
void sdpa_fused_kernel(const float* __restrict__ Q, const float* __restrict__ K,
                       const float* __restrict__ V, const int* __restrict__ M,
                       float* __restrict__ outO, float* __restrict__ outA)
{
  __shared__ __align__(16) unsigned short sK[2][32 * KROW];       // double buf
  __shared__ __align__(16) unsigned short sV[2][D_HEAD * VROW];   // double buf
  __shared__ __align__(16) unsigned short sP[WPB][16 * PROW];

  const int tid  = threadIdx.x;
  const int lane = tid & 31;
  const int wib  = tid >> 5;
  const int bid  = blockIdx.x;
  const int head = bid >> 5;                 // 32 blocks per (b,h)
  const int mt   = ((bid & 31) << 2) + wib;  // 4 Q-tiles per block, same head
  const int b    = head >> 4;
  const int half = lane >> 4;
  const int c16  = lane & 15;
  const int row0 = mt * 16;

  const float* Qb = Q + (size_t)head * S_LEN * D_HEAD;
  const float* Kb = K + (size_t)head * S_LEN * D_HEAD;
  const float* Vb = V + (size_t)head * S_LEN * D_HEAD;
  const int*   Mb = M + (size_t)b * S_LEN * S_LEN;
  float* Ob = outO + (size_t)head * S_LEN * D_HEAD;
  float* Ab = outA + (size_t)head * S_LEN * S_LEN;

  // ---- Q A-fragments (vector loads, scale*log2e folded in)
  Frag aq0, aq1;
  {
    const float* qr = Qb + (size_t)(row0 + c16) * D_HEAD;
    const int o = half << 3;
    v4f x0 = *(const v4f*)(qr + o);          v4f x1 = *(const v4f*)(qr + o + 4);
    v4f x2 = *(const v4f*)(qr + 16 + o);     v4f x3 = *(const v4f*)(qr + 16 + o + 4);
    v4f y0 = *(const v4f*)(qr + 32 + o);     v4f y1 = *(const v4f*)(qr + 32 + o + 4);
    v4f y2 = *(const v4f*)(qr + 48 + o);     v4f y3 = *(const v4f*)(qr + 48 + o + 4);
    x0 *= QK_SCALE_LOG2E; x1 *= QK_SCALE_LOG2E;
    x2 *= QK_SCALE_LOG2E; x3 *= QK_SCALE_LOG2E;
    y0 *= QK_SCALE_LOG2E; y1 *= QK_SCALE_LOG2E;
    y2 *= QK_SCALE_LOG2E; y3 *= QK_SCALE_LOG2E;
    aq0.q[0] = pack8(x0, x1); aq0.q[1] = pack8(x2, x3);
    aq1.q[0] = pack8(y0, y1); aq1.q[1] = pack8(y2, y3);
  }

  // ---- pass 1: per-lane online (max, sum-2^); branchless, no shuffles
  float ml[8], ll[8];
#pragma unroll
  for (int j = 0; j < 8; ++j) { ml[j] = MASKED; ll[j] = 0.f; }

  stage_K(Kb, sK[0], tid);                       // prime the pipeline
  for (int kt = 0; kt < NKT; ++kt) {
    const int n0 = kt * 32;
    __syncthreads();                             // buf[kt&1] staged & free
    if (kt + 1 < NKT) {
      if (kt + 2 < NKT)
        __builtin_prefetch(Kb + (size_t)(n0 + 64 + (tid >> 2)) * D_HEAD +
                               ((tid & 3) << 4), 0, 0);
      stage_K(Kb + (size_t)(n0 + 32) * D_HEAD, sK[(kt + 1) & 1], tid);
    }
    v8f s0 = {}, s1 = {};
    score_tile(sK[kt & 1], Mb, aq0.v, aq1.v, n0, row0, half, c16, s0, s1);
#pragma unroll
    for (int j = 0; j < 8; ++j) {
      float t = fmaxf(fmaxf(s0[j], s1[j]), ml[j]);      // v_max3_num_f32
      ll[j] = ll[j] * fast_exp2(ml[j] - t)
            + fast_exp2(s0[j] - t) + fast_exp2(s1[j] - t);
      ml[j] = t;
    }
  }

  // merge 16 per-lane (m,l) pairs of each row; z = m + log2(l)
  float zrow[8];
#pragma unroll
  for (int j = 0; j < 8; ++j) {
    float m = ml[j], l = ll[j];
#pragma unroll
    for (int x = 1; x < 16; x <<= 1) {
      float mo = __shfl_xor(m, x, 16);
      float lo = __shfl_xor(l, x, 16);
      float mn = fmaxf(m, mo);
      l = l * fast_exp2(m - mn) + lo * fast_exp2(mo - mn);
      m = mn;
    }
    zrow[j] = m + fast_log2(fmaxf(l, 1.0e-30f));   // p = 2^(s - z)
  }

  // ---- pass 2: recompute scores, emit final P (NT stores), O += P*V
  v8f o0 = {}, o1 = {}, o2 = {}, o3 = {};
  unsigned short* sPw = sP[wib];

  __syncthreads();                               // pass-1 reads fully done
  stage_K(Kb, sK[0], tid);
  stage_VT(Vb, sV[0], tid);
  for (int kt = 0; kt < NKT; ++kt) {
    const int n0 = kt * 32;
    __syncthreads();
    if (kt + 1 < NKT) {
      if (kt + 2 < NKT) {
        __builtin_prefetch(Kb + (size_t)(n0 + 64 + (tid >> 2)) * D_HEAD +
                               ((tid & 3) << 4), 0, 0);
        __builtin_prefetch(Vb + (size_t)(n0 + 64 + (tid >> 2)) * D_HEAD +
                               ((tid & 3) << 4), 0, 0);
      }
      stage_K(Kb + (size_t)(n0 + 32) * D_HEAD, sK[(kt + 1) & 1], tid);
      stage_VT(Vb + (size_t)(n0 + 32) * D_HEAD, sV[(kt + 1) & 1], tid);
    }

    v8f s0 = {}, s1 = {};
    score_tile(sK[kt & 1], Mb, aq0.v, aq1.v, n0, row0, half, c16, s0, s1);

#pragma unroll
    for (int j = 0; j < 8; ++j) {
      const int rl = j + 8 * half;
      float p0 = fast_exp2(s0[j] - zrow[j]);     // masked -> 2^(-huge) = 0
      float p1 = fast_exp2(s1[j] - zrow[j]);
      // compulsory write-once output: non-temporal, keep L2 for K/V/mask
      float* arow = Ab + (size_t)(row0 + rl) * S_LEN + n0 + c16;
      __builtin_nontemporal_store(p0, arow);
      __builtin_nontemporal_store(p1, arow + 16);
      // stage P as bf16 for the C-layout -> A-fragment transpose
      sPw[rl * PROW + c16]      = f2bf_bits(p0);
      sPw[rl * PROW + 16 + c16] = f2bf_bits(p1);
    }

    // P A-fragment: 2x ds_load_b128 (same-wave LDS is in-order, no barrier)
    Frag ap;
    {
      const unsigned short* pr = sPw + c16 * PROW;
      const int o = half << 3;
      ap.q[0] = *(const v4u*)(pr + o);
      ap.q[1] = *(const v4u*)(pr + 16 + o);
    }

    // V B-fragments from transposed sV: 2x ds_load_b128 each
    Frag bv0, bv1, bv2, bv3;
    {
      const int o = half << 3;
      const unsigned short* sv = sV[kt & 1];
      const unsigned short* vr0 = sv + (size_t)(0  + c16) * VROW;
      const unsigned short* vr1 = sv + (size_t)(16 + c16) * VROW;
      const unsigned short* vr2 = sv + (size_t)(32 + c16) * VROW;
      const unsigned short* vr3 = sv + (size_t)(48 + c16) * VROW;
      bv0.q[0] = *(const v4u*)(vr0 + o); bv0.q[1] = *(const v4u*)(vr0 + 16 + o);
      bv1.q[0] = *(const v4u*)(vr1 + o); bv1.q[1] = *(const v4u*)(vr1 + 16 + o);
      bv2.q[0] = *(const v4u*)(vr2 + o); bv2.q[1] = *(const v4u*)(vr2 + 16 + o);
      bv3.q[0] = *(const v4u*)(vr3 + o); bv3.q[1] = *(const v4u*)(vr3 + 16 + o);
    }
    o0 = wmma_bf16(ap.v, bv0.v, o0);
    o1 = wmma_bf16(ap.v, bv1.v, o1);
    o2 = wmma_bf16(ap.v, bv2.v, o2);
    o3 = wmma_bf16(ap.v, bv3.v, o3);
  }

  // ---- store O (normalized; write-once -> non-temporal)
#pragma unroll
  for (int j = 0; j < 8; ++j) {
    float* orow = Ob + (size_t)(row0 + j + 8 * half) * D_HEAD + c16;
    __builtin_nontemporal_store(o0[j], orow);
    __builtin_nontemporal_store(o1[j], orow + 16);
    __builtin_nontemporal_store(o2[j], orow + 32);
    __builtin_nontemporal_store(o3[j], orow + 48);
  }
}

} // namespace

extern "C" void kernel_launch(void* const* d_in, const int* in_sizes, int n_in,
                              void* d_out, int out_size, void* d_ws, size_t ws_size,
                              hipStream_t stream) {
  const float* Q = (const float*)d_in[0];
  const float* K = (const float*)d_in[1];
  const float* V = (const float*)d_in[2];
  const int*   M = (const int*)d_in[3];

  float* outO = (float*)d_out;                                    // [B,H,S,D]
  float* outA = outO + (size_t)N_BATCH * N_HEAD * S_LEN * D_HEAD; // [B,H,S,S]

  const int totalWaves = N_BATCH * N_HEAD * (S_LEN / 16);  // 4096
  dim3 grid(totalWaves / WPB);                             // 1024 blocks
  dim3 block(WPB * 32);                                    // 4 waves/block

  sdpa_fused_kernel<<<grid, block, 0, stream>>>(Q, K, V, M, outO, outA);
}
